// LinearAttention_63574105915901
// MI455X (gfx1250) — matrix operand order repaired and checked by reading, not compiled, not run on hardware
//
#include <hip/hip_runtime.h>
#include <hip/hip_bf16.h>

// ---------------------------------------------------------------------------
// Types for gfx1250 WMMA (wave32)
// ---------------------------------------------------------------------------
typedef __bf16 bf16_t;
typedef bf16_t v16bf __attribute__((ext_vector_type(16)));
typedef float  v8f   __attribute__((ext_vector_type(8)));

union FragU {
    v16bf v;
    unsigned short u[16];
    uint4 q[2];
};

__device__ __forceinline__ unsigned short f32_to_bf16(float f) {
    unsigned int u = __float_as_uint(f);
    u += 0x7FFFu + ((u >> 16) & 1u);   // round-to-nearest-even
    return (unsigned short)(u >> 16);
}

// ---------------------------------------------------------------------------
// Fragment loaders (wave32): contiguous k order k = 16*half + s, same for A and B (not ISA table order).
// All loads are 16-byte vectors: source layouts are chosen to match the
// fragment layouts exactly.
// ---------------------------------------------------------------------------

// A-matrix 16x32 bf16, row-major source [M][K] with leading dim lda.
// lanes 0-15: M=lane, K = 0..15 ; lanes 16-31: M=lane-16, K = 16..31
__device__ __forceinline__ v16bf load_a_frag(const unsigned short* src,
                                             int r0, int k0, int lda, int lane) {
    const int half = lane >> 4;
    const int m    = lane & 15;
    const unsigned short* p = src + (size_t)(r0 + m) * lda + k0 + half * 16;
    FragU f;
    f.q[0] = *(const uint4*)(p);        // K offsets half*16     .. +7
    f.q[1] = *(const uint4*)(p + 8);    // K offsets half*16 + 8 .. +15
    return f.v;
}

// B-matrix 32x16 bf16 from a TRANSPOSED source: B(k,n) = srcT[n*ldk + k].
// lanes 0-15: N=n0+lane, K = 0..15 ; lanes 16-31: N=n0+lane-16, K = 16..31
// => per lane, 16 contiguous K values: two 16-byte loads.
__device__ __forceinline__ v16bf load_bT_frag(const unsigned short* srcT,
                                              int k0, int n0, int ldk, int lane) {
    const int half = lane >> 4;
    const unsigned short* p = srcT + (size_t)(n0 + (lane & 15)) * ldk + k0 + half * 16;
    FragU f;
    f.q[0] = *(const uint4*)(p);        // K offsets half*16     .. +7
    f.q[1] = *(const uint4*)(p + 8);    // K offsets half*16 + 8 .. +15
    return f.v;
}

__device__ __forceinline__ v8f wmma_bf16(v16bf a, v16bf b, v8f c) {
    return __builtin_amdgcn_wmma_f32_16x16x32_bf16(
        /*neg_a=*/false, a, /*neg_b=*/false, b,
        /*c_mod=*/(short)0, c, /*reuse_a=*/false, /*reuse_b=*/false);
}

// ---------------------------------------------------------------------------
// Kernel: fp32 -> bf16 cast (plain)
// ---------------------------------------------------------------------------
__global__ void cast_bf16_kernel(const float* __restrict__ in,
                                 unsigned short* __restrict__ out, int n) {
    int i = blockIdx.x * blockDim.x + threadIdx.x;
    if (i < n) out[i] = f32_to_bf16(in[i]);
}

// ---------------------------------------------------------------------------
// Kernel: fp32 -> bf16 cast with transpose: in[R][C] -> out[C][R]
// ---------------------------------------------------------------------------
__global__ void cast_transpose_bf16_kernel(const float* __restrict__ in,
                                           unsigned short* __restrict__ out,
                                           int rows, int cols) {
    int idx = blockIdx.x * blockDim.x + threadIdx.x;
    if (idx < rows * cols) {
        int c = idx % cols;
        int r = idx / cols;
        out[(size_t)c * rows + r] = f32_to_bf16(in[idx]);
    }
}

// ---------------------------------------------------------------------------
// Kernel: QKV projection (64x64 tile) + per-head softmax epilogue for Q/K.
// Q written row-major [n][512]; K,V written transposed [b][512][4096] so the
// context kernel gets natural (vectorizable) fragment layouts.
// grid: (R/64, 512/64, 3)   block: 128 (4 waves, wave w -> rows w*16..+15)
// ---------------------------------------------------------------------------
__global__ __launch_bounds__(128)
void gemm_qkv_kernel(const unsigned short* __restrict__ xb,
                     const unsigned short* __restrict__ WqT,
                     const unsigned short* __restrict__ WkT,
                     const unsigned short* __restrict__ WvT,
                     unsigned short* __restrict__ Qb,
                     unsigned short* __restrict__ Kt,
                     unsigned short* __restrict__ Vt) {
    __shared__ float tile[64 * 65];   // padded: stride-65 kills bank conflicts

    const int lane = threadIdx.x & 31;
    const int wave = threadIdx.x >> 5;
    const int z    = blockIdx.z;

    const unsigned short* WT = (z == 0) ? WqT : (z == 1) ? WkT : WvT;

    const int r0 = blockIdx.x * 64 + wave * 16;   // global row of this wave's strip
    const int c0 = blockIdx.y * 64;               // global column base of tile

    v8f acc[4] = {v8f{0}, v8f{0}, v8f{0}, v8f{0}};

    for (int k0 = 0; k0 < 512; k0 += 32) {
        if (k0 + 32 < 512) {
            __builtin_prefetch((const void*)(xb + (size_t)(r0 + (lane & 15)) * 512 + k0 + 32), 0, 1);
        }
        v16bf a = load_a_frag(xb, r0, k0, 512, lane);
#pragma unroll
        for (int t = 0; t < 4; ++t) {
            v16bf b = load_bT_frag(WT, k0, c0 + t * 16, 512, lane);
            acc[t] = wmma_bf16(a, b, acc[t]);
        }
    }

    // Stage fp32 tile in LDS (C layout: VGPR i -> M = i (+8 for upper half-wave))
    const int mofs = (lane >> 4) * 8;
    const int ncol = lane & 15;
#pragma unroll
    for (int t = 0; t < 4; ++t) {
#pragma unroll
        for (int i = 0; i < 8; ++i) {
            tile[(wave * 16 + i + mofs) * 65 + t * 16 + ncol] = acc[t][i];
        }
    }
    __syncthreads();

    const int rowBase = blockIdx.x * 64;
    const int b       = rowBase >> 12;      // batch index (4096 rows per batch)
    const int nloc0   = rowBase & 4095;     // token index within batch

    if (z < 2) {
        // softmax over the 64 columns of each row (one head per tile)
        if (threadIdx.x < 64) {
            const int row = threadIdx.x;
            float mx = -3.4e38f;
#pragma unroll 8
            for (int j = 0; j < 64; ++j) mx = fmaxf(mx, tile[row * 65 + j]);
            float s = 0.0f;
#pragma unroll 8
            for (int j = 0; j < 64; ++j) {
                float e = __expf(tile[row * 65 + j] - mx);
                tile[row * 65 + j] = e;
                s += e;
            }
            const float inv = 1.0f / s;
            if (z == 0) {
                unsigned short* dst = Qb + (size_t)(rowBase + row) * 512 + c0;
#pragma unroll 8
                for (int j = 0; j < 64; ++j) dst[j] = f32_to_bf16(tile[row * 65 + j] * inv);
            } else {
                // K stored transposed: Kt[(b*512 + d) * 4096 + n]
#pragma unroll 8
                for (int j = 0; j < 64; ++j) {
                    Kt[(size_t)(b * 512 + c0 + j) * 4096 + nloc0 + row] =
                        f32_to_bf16(tile[row * 65 + j] * inv);
                }
            }
        }
    } else {
        // V stored transposed: Vt[(b*512 + d) * 4096 + n]; consecutive tid -> consecutive n
        for (int idx = threadIdx.x; idx < 4096; idx += 128) {
            const int row = idx & 63;       // n (inner -> coalesced)
            const int col = idx >> 6;       // d
            Vt[(size_t)(b * 512 + c0 + col) * 4096 + nloc0 + row] =
                f32_to_bf16(tile[row * 65 + col]);
        }
    }
}

// ---------------------------------------------------------------------------
// Kernel: context[b,h] = K^T @ V  (64 x 64, reduced over n = 4096)
// Kt is [b][512][4096] -> K^T is natural row-major A.  Vt gives contiguous-K B.
// Output stored transposed: ctxT[e][d], so gemm_head's B fragments vectorize.
// grid: 32 (b*8+h)   block: 256 (8 waves: 4 m-strips x 2 n-halves)
// ---------------------------------------------------------------------------
__global__ __launch_bounds__(256)
void ctx_kernel(const unsigned short* __restrict__ Kt,
                const unsigned short* __restrict__ Vt,
                unsigned short* __restrict__ ctxb) {
    __shared__ float cbuf[64 * 65];

    const int lane = threadIdx.x & 31;
    const int wave = threadIdx.x >> 5;
    const int bh   = blockIdx.x;
    const int b    = bh >> 3;
    const int col0 = (bh & 7) * 64;      // head's d/e offset within 512

    const unsigned short* KtB = Kt + (size_t)b * 512 * 4096;
    const unsigned short* VtB = Vt + (size_t)b * 512 * 4096;

    const int mstrip = wave & 3;         // which 16 rows (d) of the 64x64 context
    const int nhalf  = wave >> 2;        // which half of n

    v8f acc[4] = {v8f{0}, v8f{0}, v8f{0}, v8f{0}};

    const int nEnd = nhalf * 2048 + 2048;
    for (int n0 = nhalf * 2048; n0 < nEnd; n0 += 32) {
        if (n0 + 32 < nEnd) {
            __builtin_prefetch((const void*)(KtB + (size_t)(col0 + mstrip * 16 + (lane & 15)) * 4096 + n0 + 32), 0, 1);
        }
        v16bf a = load_a_frag(KtB, col0 + mstrip * 16, n0, 4096, lane);
#pragma unroll
        for (int t = 0; t < 4; ++t) {
            v16bf bfrag = load_bT_frag(VtB, n0, col0 + t * 16, 4096, lane);
            acc[t] = wmma_bf16(a, bfrag, acc[t]);
        }
    }

    for (int idx = threadIdx.x; idx < 64 * 65; idx += 256) cbuf[idx] = 0.0f;
    __syncthreads();

    const int mofs = (lane >> 4) * 8;
    const int ncol = lane & 15;
#pragma unroll
    for (int t = 0; t < 4; ++t) {
#pragma unroll
        for (int i = 0; i < 8; ++i) {
            atomicAdd(&cbuf[(mstrip * 16 + i + mofs) * 65 + t * 16 + ncol], acc[t][i]);
        }
    }
    __syncthreads();

    // store transposed: ctxT[e*64 + d]; consecutive tid -> consecutive d (coalesced),
    // cbuf read stride 65 (conflict-free)
    for (int idx = threadIdx.x; idx < 4096; idx += 256) {
        const int d = idx & 63;
        const int e = idx >> 6;
        ctxb[(size_t)bh * 4096 + e * 64 + d] = f32_to_bf16(cbuf[d * 65 + e]);
    }
}

// ---------------------------------------------------------------------------
// Kernel: per-head out = Q @ context   (K = 64; context stored as ctxT[e][d])
// grid: (R/64, 8 heads)   block: 128
// ---------------------------------------------------------------------------
__global__ __launch_bounds__(128)
void gemm_head_kernel(const unsigned short* __restrict__ Qb,
                      const unsigned short* __restrict__ ctxb,
                      unsigned short* __restrict__ O1b) {
    const int lane = threadIdx.x & 31;
    const int wave = threadIdx.x >> 5;
    const int head = blockIdx.y;
    const int r0   = blockIdx.x * 64 + wave * 16;
    const int b    = blockIdx.x >> 6;                 // 64 row-tiles per batch
    const unsigned short* C = ctxb + (size_t)(b * 8 + head) * 4096;  // ctxT[e][d]

    v8f acc[4] = {v8f{0}, v8f{0}, v8f{0}, v8f{0}};

#pragma unroll
    for (int k0 = 0; k0 < 64; k0 += 32) {
        v16bf a = load_a_frag(Qb, r0, head * 64 + k0, 512, lane);
#pragma unroll
        for (int t = 0; t < 4; ++t) {
            v16bf bfrag = load_bT_frag(C, k0, t * 16, 64, lane);
            acc[t] = wmma_bf16(a, bfrag, acc[t]);
        }
    }

    const int mofs = (lane >> 4) * 8;
    const int ncol = lane & 15;
#pragma unroll
    for (int t = 0; t < 4; ++t) {
#pragma unroll
        for (int i = 0; i < 8; ++i) {
            O1b[(size_t)(r0 + i + mofs) * 512 + head * 64 + t * 16 + ncol] =
                f32_to_bf16(acc[t][i]);
        }
    }
}

// ---------------------------------------------------------------------------
// Kernel: final = O1 @ Wo + bo   (fp32 output; Wo pre-transposed)
// grid: (R/64, 8)   block: 128
// ---------------------------------------------------------------------------
__global__ __launch_bounds__(128)
void gemm_out_kernel(const unsigned short* __restrict__ O1b,
                     const unsigned short* __restrict__ WoT,
                     const float* __restrict__ bo,
                     float* __restrict__ out) {
    const int lane = threadIdx.x & 31;
    const int wave = threadIdx.x >> 5;
    const int r0   = blockIdx.x * 64 + wave * 16;
    const int c0   = blockIdx.y * 64;

    v8f acc[4] = {v8f{0}, v8f{0}, v8f{0}, v8f{0}};

    for (int k0 = 0; k0 < 512; k0 += 32) {
        if (k0 + 32 < 512) {
            __builtin_prefetch((const void*)(O1b + (size_t)(r0 + (lane & 15)) * 512 + k0 + 32), 0, 1);
        }
        v16bf a = load_a_frag(O1b, r0, k0, 512, lane);
#pragma unroll
        for (int t = 0; t < 4; ++t) {
            v16bf bfrag = load_bT_frag(WoT, k0, c0 + t * 16, 512, lane);
            acc[t] = wmma_bf16(a, bfrag, acc[t]);
        }
    }

    const int mofs = (lane >> 4) * 8;
    const int ncol = lane & 15;
#pragma unroll
    for (int t = 0; t < 4; ++t) {
        const int col = c0 + t * 16 + ncol;
        const float bias = bo[col];
#pragma unroll
        for (int i = 0; i < 8; ++i) {
            out[(size_t)(r0 + i + mofs) * 512 + col] = acc[t][i] + bias;
        }
    }
}

// ---------------------------------------------------------------------------
// Host launcher
// ---------------------------------------------------------------------------
extern "C" void kernel_launch(void* const* d_in, const int* in_sizes, int n_in,
                              void* d_out, int out_size, void* d_ws, size_t ws_size,
                              hipStream_t stream) {
    const float* x  = (const float*)d_in[0];   // [4,4096,512]
    const float* Wq = (const float*)d_in[1];   // [512,512]
    const float* Wk = (const float*)d_in[2];
    const float* Wv = (const float*)d_in[3];
    const float* Wo = (const float*)d_in[4];
    const float* bo = (const float*)d_in[5];   // [512]

    const int R = 4 * 4096;                    // 16384 rows
    const int D = 512;
    const size_t XB_BYTES = (size_t)R * D * 2; // 16 MB
    const size_t W_BYTES  = (size_t)D * D * 2; // 512 KB

    char* base = (char*)d_ws;
    unsigned short* xb   = (unsigned short*)(base);                       // reused as O1b later
    unsigned short* WqT  = (unsigned short*)(base + XB_BYTES);
    unsigned short* WkT  = (unsigned short*)(base + XB_BYTES + 1 * W_BYTES);
    unsigned short* WvT  = (unsigned short*)(base + XB_BYTES + 2 * W_BYTES);
    unsigned short* WoT  = (unsigned short*)(base + XB_BYTES + 3 * W_BYTES);
    unsigned short* Qb   = (unsigned short*)(base + XB_BYTES + 4 * W_BYTES);
    unsigned short* Kt   = (unsigned short*)(base + 2 * XB_BYTES + 4 * W_BYTES);
    unsigned short* Vt   = (unsigned short*)(base + 3 * XB_BYTES + 4 * W_BYTES);
    unsigned short* ctxb = (unsigned short*)(base + 4 * XB_BYTES + 4 * W_BYTES);
    unsigned short* O1b  = xb;                 // x no longer needed after QKV GEMM

    // 1) fp32 -> bf16 casts; weights transposed for vectorized B fragments
    {
        int n = R * D;
        cast_bf16_kernel<<<(n + 255) / 256, 256, 0, stream>>>(x, xb, n);
        int nw = D * D;
        cast_transpose_bf16_kernel<<<(nw + 255) / 256, 256, 0, stream>>>(Wq, WqT, D, D);
        cast_transpose_bf16_kernel<<<(nw + 255) / 256, 256, 0, stream>>>(Wk, WkT, D, D);
        cast_transpose_bf16_kernel<<<(nw + 255) / 256, 256, 0, stream>>>(Wv, WvT, D, D);
        cast_transpose_bf16_kernel<<<(nw + 255) / 256, 256, 0, stream>>>(Wo, WoT, D, D);
    }

    // 2) QKV projections + per-head softmax on Q/K; K,V written transposed
    gemm_qkv_kernel<<<dim3(R / 64, D / 64, 3), 128, 0, stream>>>(
        xb, WqT, WkT, WvT, Qb, Kt, Vt);

    // 3) per-(b,h) context = K^T @ V (stored transposed as ctxT[e][d])
    ctx_kernel<<<32, 256, 0, stream>>>(Kt, Vt, ctxb);

    // 4) per-head out = Q @ context (writes into recycled x buffer)
    gemm_head_kernel<<<dim3(R / 64, 8), 128, 0, stream>>>(Qb, ctxb, O1b);

    // 5) final projection + bias -> fp32 output
    gemm_out_kernel<<<dim3(R / 64, D / 64), 128, 0, stream>>>(
        O1b, WoT, bo, (float*)d_out);
}
